// RNN_82197084111089
// MI455X (gfx1250) — matrix operand order repaired and checked by reading, not compiled, not run on hardware
//
#include <hip/hip_runtime.h>
#include <stdint.h>

// ---------------------------------------------------------------------------
// Types for CDNA5 WMMA (gfx1250, wave32)
// ---------------------------------------------------------------------------
typedef __attribute__((ext_vector_type(16))) __bf16 v16bf;
typedef __attribute__((ext_vector_type(8)))  float  v8f;

union Frag {           // 32 bytes: one bf16 A/B fragment per lane
    v16bf    bf;
    uint16_t u16[16];
    uint32_t u32[8];
    uint4    q[2];
};

__device__ __forceinline__ uint16_t f2bf(float f) {
    uint32_t u = __float_as_uint(f);
    uint32_t r = u + 0x7FFFu + ((u >> 16) & 1u);   // round-to-nearest-even
    return (uint16_t)(r >> 16);
}

__device__ __forceinline__ float fast_sig(float x) {
    // 1 / (1 + 2^(-x*log2e))  -> v_exp_f32 + v_rcp_f32
    return __builtin_amdgcn_rcpf(1.0f + __builtin_amdgcn_exp2f(-1.4426950408889634f * x));
}
__device__ __forceinline__ float fast_tanh(float x) {
    return 2.0f * fast_sig(2.0f * x) - 1.0f;
}

// ---------------------------------------------------------------------------
// Pack f32 [K,N] weight -> bf16 fragments laid out exactly as WMMA B operand:
//   dst[((kt*NT + nt)*32 + lane)*8 + v] packs (k=32kt+2v+8*(lane>>4)+8*(v>>2), n=16nt+(lane&15))
// so a B fragment is 2 contiguous b128 loads per lane.
// ---------------------------------------------------------------------------
__global__ void pack_weights(const float* __restrict__ src, uint32_t* __restrict__ dst,
                             int N, int count) {
    int q = blockIdx.x * blockDim.x + threadIdx.x;
    if (q >= count) return;
    int v    = q & 7;
    int l    = (q >> 3) & 31;
    int rest = q >> 8;
    int NT   = N >> 4;
    int nt   = rest % NT;
    int kt   = rest / NT;
    int k    = 32 * kt + 2 * v + 8 * (l >> 4) + 8 * (v >> 2);
    int n    = 16 * nt + (l & 15);
    uint32_t lo = f2bf(src[(size_t)k * N + n]);
    uint32_t hi = f2bf(src[(size_t)(k + 1) * N + n]);
    dst[q] = lo | (hi << 16);
}

// ---------------------------------------------------------------------------
// Input projection GEMM, TIME-MAJOR: logical row m' = t*1024 + b.
// out is stored in the recurrence's C-fragment layout:
//   xzp[((t*64 + btile)*64 + nt)*256 + lane*8 + r]
// so the LSTM reads each gate tile with 2 coalesced b128 loads.
// Block covers 64 rows (one t, 4 b-tiles) x 1024 cols; every B fragment from L2
// is reused by 4 WMMAs. 512 threads = 16 waves; wave w owns ntiles [4w..4w+4).
// ---------------------------------------------------------------------------
template<int KDIM, bool A_F32>
__global__ __launch_bounds__(512) void gemm_proj(
    const void*     __restrict__ A,     // A_F32: x[B=1024][T=256][KDIM] f32
                                        // else : h0seq[t][b][KDIM] bf16 (time-major)
    const uint32_t* __restrict__ Wpk,   // packed [KDIM/32][64][32][8]
    const float*    __restrict__ bias,  // [1024]
    float*          __restrict__ outpk) // packed xz, 1 GB
{
    constexpr int NKT = KDIM / 32;
    __shared__ __align__(16) uint16_t aS[64 * KDIM];

    const int tid = threadIdx.x;
    const int m0  = blockIdx.x * 64;        // m' = t*1024 + b
    const int t   = m0 >> 10;
    const int b0  = m0 & 1023;

    // stage 64 x KDIM A tile into LDS as bf16
    if (A_F32) {
        // x is [b][t][f]: row i/KDIM is batch offset, gather with stride T*KDIM
        const float* xf = (const float*)A;
        for (int i = tid; i < 64 * KDIM; i += 512) {
            int row = i / KDIM, k = i % KDIM;
            aS[i] = f2bf(xf[((size_t)(b0 + row) * 256 + t) * KDIM + k]);
        }
    } else {
        // h0seq time-major: 64 rows are contiguous memory
        const uint4* src = (const uint4*)((const uint16_t*)A + ((size_t)t * 1024 + b0) * KDIM);
        uint4* dst = (uint4*)aS;
        for (int i = tid; i < 64 * KDIM / 8; i += 512) dst[i] = src[i];
    }
    __syncthreads();

    const int w = tid >> 5, lane = tid & 31;
    const int col = lane & 15, half = lane >> 4;

    v8f acc[4][4];   // [gate j][b-tile]
    #pragma unroll
    for (int j = 0; j < 4; ++j) {
        float b = bias[16 * (4 * w + j) + col];
        #pragma unroll
        for (int mt = 0; mt < 4; ++mt)
            #pragma unroll
            for (int r = 0; r < 8; ++r) acc[j][mt][r] = b;
    }

    #pragma unroll
    for (int kt = 0; kt < NKT; ++kt) {
        Frag a[4];
        #pragma unroll
        for (int mt = 0; mt < 4; ++mt) {
            const uint16_t* ap = aS + (size_t)(16 * mt + col) * KDIM + 32 * kt + 8 * half;
            a[mt].q[0] = *(const uint4*)(ap);
            a[mt].q[1] = *(const uint4*)(ap + 16);
        }
        #pragma unroll
        for (int j = 0; j < 4; ++j) {
            Frag b;
            const uint32_t* bp = Wpk + (((size_t)kt * 64 + (4 * w + j)) * 32 + lane) * 8;
            b.q[0] = *(const uint4*)(bp);
            b.q[1] = *(const uint4*)(bp + 4);
            #pragma unroll
            for (int mt = 0; mt < 4; ++mt)
                acc[j][mt] = __builtin_amdgcn_wmma_f32_16x16x32_bf16(
                                 false, a[mt].bf, false, b.bf, (short)0, acc[j][mt], false, false);
        }
    }

    // packed store: 8 contiguous floats per lane (2 coalesced b128 stores)
    #pragma unroll
    for (int j = 0; j < 4; ++j) {
        #pragma unroll
        for (int mt = 0; mt < 4; ++mt) {
            int btile = (b0 >> 4) + mt;
            float* op = outpk + (((size_t)t * 64 + btile) * 64 + (4 * w + j)) * 256 + lane * 8;
            *(v8f*)op = acc[j][mt];
        }
    }
}

// ---------------------------------------------------------------------------
// LSTM recurrence, one layer. 64 blocks x 512 threads; block owns 16 batch rows
// (btile = blockIdx.x) for all T steps. Wave w owns hidden cols [16w,16w+16)
// (gate tiles {w, w+16, w+32, w+48}) so the cell update needs no cross-wave
// exchange. Hybrid U cache: k-tiles 0..3 in 128 VGPRs/wave, k-tiles 4..7 in
// 256 KB LDS (B frags pipelined 1 deep). xz arrives pre-packed in C-fragment
// layout (2 b128 loads per gate) and is PREFETCHED into acc for step t+1
// between the hS write and the closing barrier — zero extra registers, the
// loads overlap the barrier + next step's LDS reads.
// ---------------------------------------------------------------------------
__global__ __launch_bounds__(512) void lstm_rec(
    const float*    __restrict__ xzp,       // packed [T][64][64][256]
    const uint32_t* __restrict__ Upk,       // packed [8][64][32][8]
    uint16_t*       __restrict__ out_seq,   // time-major [T][B][256] bf16 (or null)
    float*          __restrict__ out_final, // [B][256] f32 (or null)
    int T)
{
    __shared__ __align__(16) uint16_t hS[16 * 256];           // 8 KB: h_{t-1}
    __shared__ __align__(16) uint32_t uS[4 * 64 * 32 * 8];    // 256 KB: U k-tiles 4..7

    const int tid = threadIdx.x, w = tid >> 5, lane = tid & 31;
    const int col = lane & 15, half = lane >> 4, rowbase = half * 8;
    const int btile = blockIdx.x;
    const int bRow0 = btile * 16;

    // one-time: copy high-K half of packed U into LDS (contiguous 256 KB)
    {
        const uint4* src = (const uint4*)(Upk + (size_t)4 * 64 * 32 * 8);
        uint4* dst = (uint4*)uS;
        for (int i = tid; i < 4 * 64 * 32 * 8 / 4; i += 512) dst[i] = src[i];
    }

    // registers: low-K half of U (4 kt x 4 gates = 128 VGPRs per wave)
    v16bf Uf[4][4];
    #pragma unroll
    for (int kt = 0; kt < 4; ++kt) {
        #pragma unroll
        for (int j = 0; j < 4; ++j) {
            Frag f;
            const uint32_t* p = Upk + (((size_t)kt * 64 + (w + 16 * j)) * 32 + lane) * 8;
            f.q[0] = *(const uint4*)(p);
            f.q[1] = *(const uint4*)(p + 4);
            Uf[kt][j] = f.bf;
        }
    }

    for (int i = tid; i < 16 * 256; i += 512) hS[i] = 0;
    v8f c;
    #pragma unroll
    for (int r = 0; r < 8; ++r) c[r] = 0.0f;
    v8f h;

    // prefetch xz for t = 0 directly into the accumulators
    v8f acc[4];
    #pragma unroll
    for (int j = 0; j < 4; ++j)
        acc[j] = *(const v8f*)(xzp + ((size_t)btile * 64 + (w + 16 * j)) * 256 + lane * 8);
    __syncthreads();

    for (int t = 0; t < T; ++t) {
        // z += h_{t-1} @ U : k-tiles 0..3 from register-cached U
        #pragma unroll
        for (int kt = 0; kt < 4; ++kt) {
            Frag a;
            const uint16_t* ap = hS + (size_t)col * 256 + 32 * kt + 8 * half;
            a.q[0] = *(const uint4*)(ap);
            a.q[1] = *(const uint4*)(ap + 16);
            #pragma unroll
            for (int j = 0; j < 4; ++j)
                acc[j] = __builtin_amdgcn_wmma_f32_16x16x32_bf16(
                             false, a.bf, false, Uf[kt][j], (short)0, acc[j], false, false);
        }
        // k-tiles 4..7 from LDS-cached U, B fragments pipelined 1 deep
        #pragma unroll
        for (int kt = 4; kt < 8; ++kt) {
            Frag a;
            const uint16_t* ap = hS + (size_t)col * 256 + 32 * kt + 8 * half;
            a.q[0] = *(const uint4*)(ap);
            a.q[1] = *(const uint4*)(ap + 16);
            Frag bcur;
            {
                const uint32_t* bp = uS + (((size_t)(kt - 4) * 64 + w) * 32 + lane) * 8;
                bcur.q[0] = *(const uint4*)(bp);
                bcur.q[1] = *(const uint4*)(bp + 4);
            }
            #pragma unroll
            for (int j = 0; j < 4; ++j) {
                Frag bnext;
                if (j < 3) {
                    const uint32_t* bp =
                        uS + (((size_t)(kt - 4) * 64 + (w + 16 * (j + 1))) * 32 + lane) * 8;
                    bnext.q[0] = *(const uint4*)(bp);
                    bnext.q[1] = *(const uint4*)(bp + 4);
                }
                acc[j] = __builtin_amdgcn_wmma_f32_16x16x32_bf16(
                             false, a.bf, false, bcur.bf, (short)0, acc[j], false, false);
                if (j < 3) bcur = bnext;
            }
        }

        // elementwise LSTM cell (gates i,f,g,o = acc[0..3]) — acc dead afterwards
        #pragma unroll
        for (int r = 0; r < 8; ++r) {
            float ig = fast_sig(acc[0][r]);
            float fg = fast_sig(acc[1][r]);
            float gg = fast_tanh(acc[2][r]);
            float og = fast_sig(acc[3][r]);
            float cn = fg * c[r] + ig * gg;
            c[r] = cn;
            h[r] = og * fast_tanh(cn);
        }

        __syncthreads();   // everyone done reading h_{t-1} from LDS
        #pragma unroll
        for (int r = 0; r < 8; ++r) {
            uint16_t hb = f2bf(h[r]);
            hS[(size_t)(rowbase + r) * 256 + 16 * w + col] = hb;
            if (out_seq)   // time-major: [t][b][256]
                out_seq[((size_t)t * 1024 + bRow0 + rowbase + r) * 256 + 16 * w + col] = hb;
        }
        // prefetch next step's xz into acc BEFORE the closing barrier:
        // overlaps the barrier wait and the next step's LDS A/B loads.
        if (t + 1 < T) {
            #pragma unroll
            for (int j = 0; j < 4; ++j)
                acc[j] = *(const v8f*)(xzp + (((size_t)(t + 1) * 64 + btile) * 64
                                             + (w + 16 * j)) * 256 + lane * 8);
        }
        __syncthreads();   // h_t visible to all waves
    }

    if (out_final) {
        #pragma unroll
        for (int r = 0; r < 8; ++r)
            out_final[(size_t)(bRow0 + rowbase + r) * 256 + 16 * w + col] = h[r];
    }
}

// ---------------------------------------------------------------------------
// Head: logits = h1 @ Wout + bout ; softmax ; capped-simplex water-filling.
// One block per batch row (tiny work, f32 VALU).
// ---------------------------------------------------------------------------
__global__ void head_kernel(const float* __restrict__ h1, const float* __restrict__ Wout,
                            const float* __restrict__ bout, float* __restrict__ out)
{
    __shared__ float lg[50];
    const int b = blockIdx.x, t = threadIdx.x;
    if (t < 50) {
        float s = bout[t];
        for (int k = 0; k < 256; ++k) s += h1[(size_t)b * 256 + k] * Wout[(size_t)k * 50 + t];
        lg[t] = s;
    }
    __syncthreads();
    if (t == 0) {
        const float UB = 0.1f;
        float mx = lg[0];
        for (int j = 1; j < 50; ++j) mx = fmaxf(mx, lg[j]);
        float sum = 0.0f;
        for (int j = 0; j < 50; ++j) { float e = __expf(lg[j] - mx); lg[j] = e; sum += e; }
        float inv = 1.0f / sum;
        float oldv[50], wc[50];
        for (int j = 0; j < 50; ++j) {
            float v = lg[j] * inv;
            oldv[j] = v;
            wc[j]   = fminf(fmaxf(v, 0.0f), UB);
        }
        bool done = false;
        for (int it = 0; it < 50 && !done; ++it) {
            float leftover = 0.0f, denom = 0.0f;
            for (int j = 0; j < 50; ++j) {
                leftover += oldv[j] - wc[j];
                if (wc[j] != UB) denom += wc[j];
            }
            float dd = (denom == 0.0f) ? 1.0f : denom;
            bool any = false;
            for (int j = 0; j < 50; ++j) {
                float wn = wc[j];
                if (wc[j] != UB) wn = wc[j] + leftover * wc[j] / dd;
                oldv[j] = wn;
                wc[j]   = wn;
                if (wn > UB) any = true;
            }
            if (!any) done = true;
            else for (int j = 0; j < 50; ++j) wc[j] = fminf(fmaxf(wc[j], 0.0f), UB);
        }
        for (int j = 0; j < 50; ++j) out[(size_t)b * 50 + j] = wc[j];
    }
}

// ---------------------------------------------------------------------------
// Host-side orchestration
// ---------------------------------------------------------------------------
extern "C" void kernel_launch(void* const* d_in, const int* in_sizes, int n_in,
                              void* d_out, int out_size, void* d_ws, size_t ws_size,
                              hipStream_t stream) {
    (void)in_sizes; (void)n_in; (void)out_size; (void)ws_size;
    const int B = 1024, T = 256, F = 128, H = 256;
    const float* x    = (const float*)d_in[0];
    const float* W0   = (const float*)d_in[1];
    const float* U0   = (const float*)d_in[2];
    const float* b0   = (const float*)d_in[3];
    const float* W1   = (const float*)d_in[4];
    const float* U1   = (const float*)d_in[5];
    const float* b1   = (const float*)d_in[6];
    const float* Wout = (const float*)d_in[7];
    const float* bout = (const float*)d_in[8];
    float* out = (float*)d_out;

    // workspace layout (all offsets 256B-aligned)
    char* ws = (char*)d_ws;
    size_t off = 0;
    float*    xzp   = (float*)   (ws + off); off += (size_t)B * T * 4 * H * 4; // 1 GB packed (reused)
    uint16_t* h0seq = (uint16_t*)(ws + off); off += (size_t)B * T * H * 2;     // 128 MB time-major
    uint32_t* W0pk  = (uint32_t*)(ws + off); off += (size_t)F * 4 * H / 2 * 4;
    uint32_t* U0pk  = (uint32_t*)(ws + off); off += (size_t)H * 4 * H / 2 * 4;
    uint32_t* W1pk  = (uint32_t*)(ws + off); off += (size_t)H * 4 * H / 2 * 4;
    uint32_t* U1pk  = (uint32_t*)(ws + off); off += (size_t)H * 4 * H / 2 * 4;
    float*    h1    = (float*)   (ws + off); off += (size_t)B * H * 4;

    // 1. swizzle weights into WMMA B-fragment layout (bf16)
    {
        int c0 = F * 4 * H / 2, c1 = H * 4 * H / 2;
        pack_weights<<<(c0 + 255) / 256, 256, 0, stream>>>(W0, W0pk, 4 * H, c0);
        pack_weights<<<(c1 + 255) / 256, 256, 0, stream>>>(U0, U0pk, 4 * H, c1);
        pack_weights<<<(c1 + 255) / 256, 256, 0, stream>>>(W1, W1pk, 4 * H, c1);
        pack_weights<<<(c1 + 255) / 256, 256, 0, stream>>>(U1, U1pk, 4 * H, c1);
    }

    const int Mblocks = B * T / 64;   // 4096 time-major blocks

    // 2. xz0 = x @ W0 + b0   (packed, time-major)
    gemm_proj<128, true ><<<Mblocks, 512, 0, stream>>>((const void*)x, W0pk, b0, xzp);
    // 3. layer-0 recurrence -> h0 sequence (bf16, time-major)
    lstm_rec<<<B / 16, 512, 0, stream>>>(xzp, U0pk, h0seq, nullptr, T);
    // 4. xz1 = h0 @ W1 + b1 (packed; reuses xz buffer)
    gemm_proj<256, false><<<Mblocks, 512, 0, stream>>>((const void*)h0seq, W1pk, b1, xzp);
    // 5. layer-1 recurrence -> final h1 (f32)
    lstm_rec<<<B / 16, 512, 0, stream>>>(xzp, U1pk, nullptr, h1, T);
    // 6. dense head + softmax + water-filling rebalance
    head_kernel<<<B, 64, 0, stream>>>(h1, Wout, bout, out);
}
